// LSTM_LSTM_27994596836266
// MI455X (gfx1250) — matrix operand order repaired and checked by reading, not compiled, not run on hardware
//
#include <hip/hip_runtime.h>
#include <hip/hip_bf16.h>
#include <math.h>

typedef __bf16 bf16;
typedef __attribute__((ext_vector_type(16))) __bf16 v16bf;
typedef __attribute__((ext_vector_type(8)))  __bf16 v8bf;
typedef __attribute__((ext_vector_type(8)))  float  v8f;
typedef __attribute__((ext_vector_type(4)))  unsigned int v4u;
typedef __attribute__((ext_vector_type(8)))  int v8i;
typedef __attribute__((ext_vector_type(4)))  int v4i;

#define B_   256
#define T_   1000
#define H_   64
#define G_   256   // 4*H
#define D2_  128   // 2*H
#define DIN_ 22

// ---- WMMA fragment loaders (CDNA5 16x16x32 bf16 layouts, ISA 7.12.2) ----

__device__ __forceinline__ v16bf load_frag16(const bf16* p0, const bf16* p1) {
  v8bf a = *(const v8bf*)p0;
  v8bf b = *(const v8bf*)p1;
  v16bf r;
#pragma unroll
  for (int i = 0; i < 8; ++i) { r[i] = a[i]; r[i + 8] = b[i]; }
  return r;
}

// A matrix 16x32 (MxK), row-major source with stride lda.
__device__ __forceinline__ v16bf load_A(const bf16* base, int lda, int m0, int k0, int lane) {
  int m  = m0 + (lane & 15);
  int kh = (lane >> 4) * 8;
  const bf16* r = base + (size_t)m * lda + k0 + kh;
  return load_frag16(r, r + 16);
}

// B matrix 32x16 (KxN) where B[k][n] = W[n][k]; W row-major with stride ldw.
__device__ __forceinline__ v16bf load_B(const bf16* W, int ldw, int n0, int k0, int lane) {
  int n  = n0 + (lane & 15);
  int kb = k0 + (lane >> 4) * 8;
  const bf16* r = W + (size_t)n * ldw + kb;
  return load_frag16(r, r + 16);
}

__device__ __forceinline__ float sigmoidf_(float x) {
  return 1.0f / (1.0f + __expf(-x));
}

// ---- Tensor Data Mover: 2D tile load (rows of 256 bf16, strided) -> LDS ----
// D# bitfields per ISA 8.3/8.4. This toolchain's 6-arg builtin form.
__device__ __forceinline__ void tdm_load_tile16x256(const bf16* gptr, bf16* lptr) {
  unsigned long long ga = (unsigned long long)(uintptr_t)gptr;
  unsigned int la = (unsigned int)(uintptr_t)lptr;  // LDS aperture: low 32 bits = LDS byte addr
  v4u g0;
  g0[0] = 1u;                                              // count=1 (valid user descriptor)
  g0[1] = la;                                              // lds_addr
  g0[2] = (unsigned int)ga;                                // global_addr[31:0]
  g0[3] = (unsigned int)((ga >> 32) & 0x1FFFFFFu) | (2u << 30);  // global_addr[56:32] | type=2
  const unsigned int td0 = G_;                             // tensor_dim0 (elements)
  const unsigned int td1 = 16u;                            // tensor_dim1
  const unsigned long long st0 = (unsigned long long)T_ * G_;  // dim0 stride (elements)
  v8i g1;
  g1[0] = (int)(1u << 16);                                 // workgroup_mask=0, data_size=1 (2B)
  g1[1] = (int)((td0 & 0xFFFFu) << 16);                    // tensor_dim0[15:0]
  g1[2] = (int)((td0 >> 16) | ((td1 & 0xFFFFu) << 16));    // td0[31:16] | td1[15:0]
  g1[3] = (int)((td1 >> 16) | ((unsigned)G_ << 16));       // td1[31:16] | tile_dim0=256
  g1[4] = (int)16u;                                        // tile_dim1=16, tile_dim2=0
  g1[5] = (int)(unsigned int)st0;                          // tensor_dim0_stride[31:0]
  g1[6] = (int)((unsigned int)(st0 >> 32) & 0xFFFFu);      // stride[47:32] | dim1_stride=0
  g1[7] = 0;
  v4i z4 = {0, 0, 0, 0};
  v8i z8 = {0, 0, 0, 0, 0, 0, 0, 0};
  __builtin_amdgcn_tensor_load_to_lds(g0, g1, z4, z4, z8, 0);
}

// ---------------------------------------------------------------------------
// Layer-0 input projection: K=22, memory-bound, plain VALU. Output bf16.
// ---------------------------------------------------------------------------
__global__ __launch_bounds__(256)
void lstm_proj_first_kernel(const float* __restrict__ x,           // (B,22,T)
                            const float* __restrict__ w_ih_first,  // (2,256,22)
                            const float* __restrict__ b_ih,        // (4,2,256)
                            const float* __restrict__ b_hh,
                            bf16* __restrict__ xw_f, bf16* __restrict__ xw_b) {
  __shared__ float sX[DIN_ * 8];
  __shared__ float sWih[2 * G_ * DIN_];
  const int tid = threadIdx.x;
  const int b   = blockIdx.x / 125;
  const int t0  = (blockIdx.x % 125) * 8;

  for (int i = tid; i < 2 * G_ * DIN_; i += blockDim.x) sWih[i] = w_ih_first[i];
  for (int i = tid; i < DIN_ * 8; i += blockDim.x) {
    int k = i >> 3, it = i & 7;
    sX[i] = x[((size_t)b * DIN_ + k) * T_ + t0 + it];
  }
  __syncthreads();

  const int g = tid;
#pragma unroll
  for (int d = 0; d < 2; ++d) {
    bf16* xw = d ? xw_b : xw_f;
    float bias = b_ih[d * G_ + g] + b_hh[d * G_ + g];
    const float* wrow = &sWih[(d * G_ + g) * DIN_];
    for (int it = 0; it < 8; ++it) {
      float s = bias;
#pragma unroll
      for (int k = 0; k < DIN_; ++k) s += sX[k * 8 + it] * wrow[k];
      xw[((size_t)b * T_ + t0 + it) * G_ + g] = (bf16)s;
    }
  }
}

// ---------------------------------------------------------------------------
// Layers 1..3 input projection: big WMMA GEMM (M=256000, N=256, K=128).
// ---------------------------------------------------------------------------
__global__ __launch_bounds__(128)
void lstm_proj_kernel(const bf16* __restrict__ in,          // (B*T, 128) bf16
                      const float* __restrict__ w_ih_rest,  // (3,2,256,128)
                      const float* __restrict__ b_ih, const float* __restrict__ b_hh,
                      bf16* __restrict__ xw_f, bf16* __restrict__ xw_b,
                      int layer /* 1..3 */) {
  __shared__ bf16 sW[128 * D2_];  // 32KB
  const int tid = threadIdx.x, lane = tid & 31, wid = tid >> 5;
  const int dir = blockIdx.y;
  const int nh  = blockIdx.z;
  const size_t row0 = (size_t)blockIdx.x * 64;
  bf16* xw = dir ? xw_b : xw_f;

  const float* wsrc = w_ih_rest + ((size_t)((layer - 1) * 2 + dir)) * G_ * D2_ +
                      (size_t)nh * 128 * D2_;
  for (int i = tid; i < 128 * D2_; i += blockDim.x) sW[i] = (bf16)wsrc[i];
  __syncthreads();

  const bf16* abase = in + row0 * D2_;
  v16bf a[4];
#pragma unroll
  for (int kk = 0; kk < 4; ++kk) a[kk] = load_A(abase, D2_, wid * 16, kk * 32, lane);

  const float* bi = b_ih + (size_t)(layer * 2 + dir) * G_ + nh * 128;
  const float* bh = b_hh + (size_t)(layer * 2 + dir) * G_ + nh * 128;

#pragma unroll 2
  for (int nt = 0; nt < 8; ++nt) {
    v8f c = {};
#pragma unroll
    for (int kk = 0; kk < 4; ++kk) {
      v16bf bfr = load_B(sW, D2_, nt * 16, kk * 32, lane);
      c = __builtin_amdgcn_wmma_f32_16x16x32_bf16(false, a[kk], false, bfr,
                                                  (short)0, c, false, false);
    }
    int nl   = nt * 16 + (lane & 15);
    int gcol = nh * 128 + nl;
    float bias = bi[nl] + bh[nl];
    int mb = wid * 16 + (lane >> 4) * 8;
#pragma unroll
    for (int v = 0; v < 8; ++v) {
      size_t row = row0 + mb + v;
      xw[row * G_ + gcol] = (bf16)(c[v] + bias);
    }
  }
}

// ---------------------------------------------------------------------------
// Persistent recurrence: one block per (16-batch tile, direction).
// Wave w owns N-tiles {w, w+4, w+8, w+12} => its accumulators hold i/f/g/o
// for hidden columns j in [16w,16w+16), so the LSTM cell runs register-local:
// no gate staging in LDS, c state in VGPRs, 2 barriers/step.
// xw tiles are double-buffered in LDS via the Tensor Data Mover, issued one
// step ahead by wave 0 and fenced with s_wait_tensorcnt.
// ---------------------------------------------------------------------------
__global__ __launch_bounds__(128)
void lstm_rec_kernel(const bf16* __restrict__ xw_f, const bf16* __restrict__ xw_b,
                     const float* __restrict__ w_hh_all,  // (4,2,256,64)
                     bf16* __restrict__ act_out,          // (B,T,128)
                     int layer) {
  __shared__ bf16 sW[G_ * H_];          // 32KB recurrent weights (bf16)
  __shared__ bf16 sH[16 * H_];          // 2KB  h state (WMMA A operand)
  __shared__ bf16 sXW[2][16 * G_];      // 16KB xw tile double buffer (TDM dest)

  const int tid = threadIdx.x, lane = tid & 31, wid = tid >> 5;
  const int dir = blockIdx.y;
  const int b0  = blockIdx.x * 16;
  const bf16* xw = dir ? xw_b : xw_f;

  const float* wsrc = w_hh_all + ((size_t)(layer * 2 + dir)) * G_ * H_;
  for (int i = tid; i < G_ * H_; i += blockDim.x) sW[i] = (bf16)wsrc[i];
  for (int i = tid; i < 16 * H_; i += blockDim.x) sH[i] = (bf16)0.0f;

  // prefetch first xw tile via TDM (wave 0 only; TENSORcnt is per-wave)
  const int t_first = dir ? (T_ - 1) : 0;
  if (wid == 0) {
    tdm_load_tile16x256(xw + ((size_t)b0 * T_ + t_first) * G_, &sXW[0][0]);
  }
  __syncthreads();

  const int l15 = lane & 15;
  const int kh8 = (lane >> 4) * 8;
  const int j   = wid * 16 + l15;     // hidden column owned by this lane
  float creg[8];
#pragma unroll
  for (int v = 0; v < 8; ++v) creg[v] = 0.0f;

  for (int tt = 0; tt < T_; ++tt) {
    const int t   = dir ? (T_ - 1 - tt) : tt;
    const int cur = tt & 1;

    // (1) recurrent GEMM: acc[gate] = h @ w_hh^T tile, gate-aligned per wave
    v16bf a0 = load_A(sH, H_, 0, 0, lane);
    v16bf a1 = load_A(sH, H_, 0, 32, lane);
    v8f acc[4];
#pragma unroll
    for (int gate = 0; gate < 4; ++gate) {
      int n0 = gate * 64 + wid * 16;
      v16bf bf0 = load_B(sW, H_, n0, 0, lane);
      v16bf bf1 = load_B(sW, H_, n0, 32, lane);
      v8f c = {};
      c = __builtin_amdgcn_wmma_f32_16x16x32_bf16(false, a0, false, bf0, (short)0, c, false, false);
      c = __builtin_amdgcn_wmma_f32_16x16x32_bf16(false, a1, false, bf1, (short)0, c, false, false);
      acc[gate] = c;
    }

    // (2) wave 0: issue TDM for next tile, then fence current (in-order per wave)
    if (wid == 0) {
      if (tt + 1 < T_) {
        int tn = dir ? (T_ - 2 - tt) : (tt + 1);
        tdm_load_tile16x256(xw + ((size_t)b0 * T_ + tn) * G_, &sXW[cur ^ 1][0]);
        __builtin_amdgcn_s_wait_tensorcnt(1);  // older (current) tile resident
      } else {
        __builtin_amdgcn_s_wait_tensorcnt(0);
      }
    }
    __syncthreads();  // tile visible; all sH fragment reads complete

    // (3) register-local LSTM cell for this lane's 8 (m, j) pairs
#pragma unroll
    for (int v = 0; v < 8; ++v) {
      int m = v + kh8;
      const bf16* gb = &sXW[cur][m * G_];
      float gi = acc[0][v] + (float)gb[j];
      float gf = acc[1][v] + (float)gb[64 + j];
      float gg = acc[2][v] + (float)gb[128 + j];
      float go = acc[3][v] + (float)gb[192 + j];
      float c_ = sigmoidf_(gf) * creg[v] + sigmoidf_(gi) * tanhf(gg);
      float h_ = sigmoidf_(go) * tanhf(c_);
      creg[v] = c_;
      sH[m * H_ + j] = (bf16)h_;
      act_out[((size_t)(b0 + m) * T_ + t) * D2_ + dir * H_ + j] = (bf16)h_;
    }
    __syncthreads();  // h update visible to next step's WMMA
  }
}

// ---------------------------------------------------------------------------
// Head: out[b] = h_final[b, T-1, :] @ w_lin^T + b_lin
// ---------------------------------------------------------------------------
__global__ __launch_bounds__(64)
void lstm_head_kernel(const bf16* __restrict__ act, const float* __restrict__ w_lin,
                      const float* __restrict__ b_lin, float* __restrict__ out) {
  const int b = blockIdx.x, o = threadIdx.x;
  if (o >= 54) return;
  const bf16* h = act + ((size_t)b * T_ + (T_ - 1)) * D2_;
  float s = b_lin[o];
#pragma unroll 4
  for (int k = 0; k < D2_; ++k) s += (float)h[k] * w_lin[o * D2_ + k];
  out[b * 54 + o] = s;
}

// ---------------------------------------------------------------------------
extern "C" void kernel_launch(void* const* d_in, const int* in_sizes, int n_in,
                              void* d_out, int out_size, void* d_ws, size_t ws_size,
                              hipStream_t stream) {
  (void)in_sizes; (void)n_in; (void)out_size; (void)ws_size;
  const float* x          = (const float*)d_in[0];
  const float* w_ih_first = (const float*)d_in[1];
  const float* w_ih_rest  = (const float*)d_in[2];
  const float* w_hh       = (const float*)d_in[3];
  const float* b_ih       = (const float*)d_in[4];
  const float* b_hh       = (const float*)d_in[5];
  const float* w_lin      = (const float*)d_in[6];
  const float* b_lin      = (const float*)d_in[7];
  float* out = (float*)d_out;

  const size_t SZ_XW  = (size_t)B_ * T_ * G_;
  const size_t SZ_ACT = (size_t)B_ * T_ * D2_;
  bf16* xw_f = (bf16*)d_ws;
  bf16* xw_b = xw_f + SZ_XW;
  bf16* act0 = xw_b + SZ_XW;
  bf16* act1 = act0 + SZ_ACT;
  bf16* acts[2] = {act0, act1};

  lstm_proj_first_kernel<<<dim3(B_ * 125), 256, 0, stream>>>(x, w_ih_first, b_ih, b_hh,
                                                             xw_f, xw_b);
  lstm_rec_kernel<<<dim3(B_ / 16, 2), 128, 0, stream>>>(xw_f, xw_b, w_hh, act0, 0);

  for (int l = 1; l <= 3; ++l) {
    bf16* ain  = acts[(l + 1) & 1];
    bf16* aout = acts[l & 1];
    lstm_proj_kernel<<<dim3((B_ * T_) / 64, 2, 2), 128, 0, stream>>>(
        ain, w_ih_rest, b_ih, b_hh, xw_f, xw_b, l);
    lstm_rec_kernel<<<dim3(B_ / 16, 2), 128, 0, stream>>>(xw_f, xw_b, w_hh, aout, l);
  }

  lstm_head_kernel<<<dim3(B_), 64, 0, stream>>>(acts[1], w_lin, b_lin, out);
}
